// AttnBlockpp_54752243089551
// MI455X (gfx1250) — compile-verified, matrix operand order checked by reading, not run on hardware
//
#include <hip/hip_runtime.h>
#include <hip/hip_bf16.h>
#include <stdint.h>

typedef _Float16 v16h __attribute__((ext_vector_type(16)));
typedef float    v8f  __attribute__((ext_vector_type(8)));
typedef _Float16 h2   __attribute__((ext_vector_type(2)));

#define BB   32
#define CC   512
#define NPIX 1024
#define CPG  16
#define EPSV 1e-5f

// Load one 16-lane-row fragment (16 f16 K-values) following the CDNA5 16-bit
// A/B VGPR layout: VGPR j holds K pair {2j, 2j+1} (+8 for j>=4), upper lane
// half (lanes 16-31) offset by +8 in K. Pairs are contiguous -> dword loads
// (the compiler coalesces the two 16-byte runs into b128 loads).
__device__ __forceinline__ v16h load_frag_row(const _Float16* rowp, int half) {
  v16h out;
#pragma unroll
  for (int j = 0; j < 8; ++j) {
    int k = 2 * j + ((j >= 4) ? 8 : 0) + half * 8;
    h2 p = *(const h2*)(rowp + k);
    out[2 * j]     = p[0];
    out[2 * j + 1] = p[1];
  }
  return out;
}

// Async-stage a contiguous 16KB (16 rows x 512 f16) tile from global to LDS.
// 256 threads x 4 iters x 16B, ASYNCcnt-tracked (CDNA5 async DMA path).
__device__ __forceinline__ void stage_tile_async(const _Float16* gsrc,
                                                 _Float16* lds_dst, int tid) {
  unsigned l0 = (unsigned)(uintptr_t)(void*)lds_dst;
#pragma unroll
  for (int it = 0; it < 4; ++it) {
    int e = (it * 256 + tid) * 8;                      // f16 elements
    unsigned dst = l0 + (unsigned)(e * 2);             // LDS byte address
    unsigned long long src = (unsigned long long)(uintptr_t)(gsrc + e);
    asm volatile("global_load_async_to_lds_b128 %0, %1, off"
                 :: "v"(dst), "v"(src) : "memory");
  }
  asm volatile("s_wait_asynccnt 0x0" ::: "memory");
  __syncthreads();
}

// -------- GroupNorm: (B,512,1024) f32 -> normalized f16 in (b, p, c) layout
__global__ __launch_bounds__(256) void gn_kernel(const float* __restrict__ x,
                                                 const float* __restrict__ gamma,
                                                 const float* __restrict__ beta,
                                                 _Float16* __restrict__ hn) {
  __shared__ float rs[256], rss[256];
  int tid = threadIdx.x;
  int b = blockIdx.x >> 5, g = blockIdx.x & 31;
  const float* xb = x + (size_t)(b * CC + g * CPG) * NPIX;
  float s = 0.f, ss = 0.f;
  for (int i = tid; i < CPG * NPIX; i += 256) { float v = xb[i]; s += v; ss += v * v; }
  rs[tid] = s; rss[tid] = ss;
  __syncthreads();
  for (int st = 128; st > 0; st >>= 1) {
    if (tid < st) { rs[tid] += rs[tid + st]; rss[tid] += rss[tid + st]; }
    __syncthreads();
  }
  float mean = rs[0] * (1.f / 16384.f);
  float var  = rss[0] * (1.f / 16384.f) - mean * mean;
  float inv  = rsqrtf(var + EPSV);
  for (int i = tid; i < CPG * NPIX; i += 256) {
    int c = i >> 10, p = i & 1023;
    int ch = g * CPG + c;
    float v = (xb[i] - mean) * inv * gamma[ch] + beta[ch];
    hn[(size_t)(b * NPIX + p) * CC + ch] = (_Float16)v;
  }
}

// -------- Convert the four 512x512 f32 weights to f16 (concatenated)
__global__ __launch_bounds__(256) void wconv_kernel(const float* __restrict__ wq,
                                                    const float* __restrict__ wk,
                                                    const float* __restrict__ wv,
                                                    const float* __restrict__ wp,
                                                    _Float16* __restrict__ w16) {
  int i = blockIdx.x * 256 + threadIdx.x;   // 0 .. 4*262144-1
  int which = i >> 18, off = i & 262143;
  const float* src = (which == 0) ? wq : (which == 1) ? wk : (which == 2) ? wv : wp;
  w16[i] = (_Float16)src[off];
}

// -------- QKV: q[p][co], k[p][co] (row = pixel), v_t[co][p] (row = channel)
// 8 waves/block share one pixel A-tile -> async-stage it to LDS once.
__global__ __launch_bounds__(256) void qkv_kernel(const _Float16* __restrict__ hn,
                                                  const _Float16* __restrict__ w16,
                                                  const float* __restrict__ bq,
                                                  const float* __restrict__ bk,
                                                  const float* __restrict__ bv,
                                                  _Float16* __restrict__ q,
                                                  _Float16* __restrict__ k,
                                                  _Float16* __restrict__ vt) {
  __shared__ _Float16 as[16 * CC];                    // 16 KB shared A tile
  int tid  = threadIdx.x;
  int wid0 = blockIdx.x * 8;
  int b  = wid0 >> 11;
  int mt = (wid0 & 2047) >> 5;                        // uniform across block
  stage_tile_async(hn + (size_t)(b * NPIX + mt * 16) * CC, as, tid);

  int wid  = wid0 + (tid >> 5);
  int lane = tid & 31, half = lane >> 4, l16 = lane & 15;
  int nt = wid & 31;

  const _Float16* arow = as + (size_t)l16 * CC;       // LDS A fragments
  const _Float16* wqr  = w16 + (size_t)(nt * 16 + l16) * CC;
  const _Float16* wkr  = wqr + 262144;
  const _Float16* wvr  = wqr + 524288;

  v8f aq = {}, ak = {}, av = {};
  for (int ki = 0; ki < 16; ++ki) {
    int ko = ki * 32;
    v16h af = load_frag_row(arow + ko, half);
    v16h bfq = load_frag_row(wqr + ko, half);
    aq = __builtin_amdgcn_wmma_f32_16x16x32_f16(false, af, false, bfq, (short)0, aq, false, false);
    v16h bfk = load_frag_row(wkr + ko, half);
    ak = __builtin_amdgcn_wmma_f32_16x16x32_f16(false, af, false, bfk, (short)0, ak, false, false);
    v16h bfv = load_frag_row(wvr + ko, half);
    av = __builtin_amdgcn_wmma_f32_16x16x32_f16(false, af, false, bfv, (short)0, av, false, false);
  }
  int co = nt * 16 + l16;
  float biasq = bq[co], biask = bk[co], biasv = bv[co];
#pragma unroll
  for (int r = 0; r < 8; ++r) {
    int p = mt * 16 + r + 8 * half;
    q[(size_t)(b * NPIX + p) * CC + co]  = (_Float16)(aq[r] + biasq);
    k[(size_t)(b * NPIX + p) * CC + co]  = (_Float16)(ak[r] + biask);
    vt[(size_t)(b * CC + co) * NPIX + p] = (_Float16)(av[r] + biasv);
  }
}

// -------- Attention: per (batch, 16-query tile) block, scores+softmax in LDS
__global__ __launch_bounds__(256) void attn_kernel(const _Float16* __restrict__ q,
                                                   const _Float16* __restrict__ k,
                                                   const _Float16* __restrict__ vt,
                                                   _Float16* __restrict__ attn) {
  extern __shared__ char smem[];
  float*    S   = (float*)smem;                        // 16*1024 f32 = 64 KB
  _Float16* P   = (_Float16*)(smem + 65536);           // 16*1024 f16 = 32 KB
  _Float16* qs  = P;                                   // q stage (16 KB) overlaps P:
                                                       // qs dead before P is written
  float*    red = (float*)(smem + 65536 + 32768);      // 16*16 f32

  int tid = threadIdx.x;
  int w = tid >> 5, lane = tid & 31, half = lane >> 4, l16 = lane & 15;
  int b = blockIdx.x >> 6, qt = blockIdx.x & 63;
  const float scale = 0.044194173824159216f;           // 512^-0.5

  // Async-stage the shared q tile (16x512 f16, contiguous) into LDS.
  stage_tile_async(q + (size_t)(b * NPIX + qt * 16) * CC, qs, tid);
  const _Float16* qrow = qs + (size_t)l16 * CC;

  // Phase 1: S[16][1024] = (q_tile · k^T) * scale, 8 key-tiles per wave
  for (int ntb = 0; ntb < 8; ++ntb) {
    int nt = w + ntb * 8;
    const _Float16* krow = k + (size_t)(b * NPIX + nt * 16 + l16) * CC;
    v8f acc = {};
    for (int ki = 0; ki < 16; ++ki) {
      v16h af = load_frag_row(qrow + ki * 32, half);
      v16h bf = load_frag_row(krow + ki * 32, half);
      acc = __builtin_amdgcn_wmma_f32_16x16x32_f16(false, af, false, bf, (short)0, acc, false, false);
    }
#pragma unroll
    for (int r = 0; r < 8; ++r) {
      int m = r + 8 * half;
      int n = nt * 16 + l16;
      S[m * NPIX + n] = acc[r] * scale;
    }
  }
  __syncthreads();

  // Phase 2: softmax per row (16 threads per row)
  {
    int m = tid >> 4, c0 = tid & 15;
    float mx = -3.4e38f;
    for (int i = 0; i < 64; ++i) mx = fmaxf(mx, S[m * NPIX + c0 + i * 16]);
    red[m * 16 + c0] = mx;
    __syncthreads();
    float rmax = red[m * 16];
#pragma unroll
    for (int t = 1; t < 16; ++t) rmax = fmaxf(rmax, red[m * 16 + t]);
    __syncthreads();
    float sum = 0.f;
    for (int i = 0; i < 64; ++i) {
      float e = __expf(S[m * NPIX + c0 + i * 16] - rmax);
      S[m * NPIX + c0 + i * 16] = e;
      sum += e;
    }
    red[m * 16 + c0] = sum;
    __syncthreads();
    float rsum = 0.f;
#pragma unroll
    for (int t = 0; t < 16; ++t) rsum += red[m * 16 + t];
    float inv = 1.f / rsum;
    for (int i = 0; i < 64; ++i)
      P[m * NPIX + c0 + i * 16] = (_Float16)(S[m * NPIX + c0 + i * 16] * inv);
  }
  __syncthreads();

  // Phase 3: out_tile = P(16x1024) x V^T, 4 channel-tiles per wave
  const _Float16* prow = P + (size_t)l16 * NPIX;
  for (int ctb = 0; ctb < 4; ++ctb) {
    int ct = w + ctb * 8;
    const _Float16* vrow = vt + (size_t)(b * CC + ct * 16 + l16) * NPIX;
    v8f acc = {};
    for (int kk = 0; kk < 32; ++kk) {
      v16h af = load_frag_row(prow + kk * 32, half);
      v16h bf = load_frag_row(vrow + kk * 32, half);
      acc = __builtin_amdgcn_wmma_f32_16x16x32_f16(false, af, false, bf, (short)0, acc, false, false);
    }
#pragma unroll
    for (int r = 0; r < 8; ++r) {
      int p = qt * 16 + r + 8 * half;
      int c = ct * 16 + l16;
      attn[(size_t)(b * NPIX + p) * CC + c] = (_Float16)acc[r];
    }
  }
}

// -------- Proj + bias + residual (f32 output, (b, c, h, w))
__global__ __launch_bounds__(256) void proj_kernel(const _Float16* __restrict__ attn,
                                                   const _Float16* __restrict__ wp16,
                                                   const float* __restrict__ bp,
                                                   const float* __restrict__ x,
                                                   float* __restrict__ out) {
  __shared__ _Float16 as[16 * CC];                    // 16 KB shared A tile
  int tid  = threadIdx.x;
  int wid0 = blockIdx.x * 8;
  int b  = wid0 >> 11;
  int mt = (wid0 & 2047) >> 5;
  stage_tile_async(attn + (size_t)(b * NPIX + mt * 16) * CC, as, tid);

  int wid  = wid0 + (tid >> 5);
  int lane = tid & 31, half = lane >> 4, l16 = lane & 15;
  int nt = wid & 31;

  const _Float16* arow = as + (size_t)l16 * CC;
  const _Float16* wrow = wp16 + (size_t)(nt * 16 + l16) * CC;
  v8f acc = {};
  for (int ki = 0; ki < 16; ++ki) {
    v16h af = load_frag_row(arow + ki * 32, half);
    v16h bf = load_frag_row(wrow + ki * 32, half);
    acc = __builtin_amdgcn_wmma_f32_16x16x32_f16(false, af, false, bf, (short)0, acc, false, false);
  }
  int co = nt * 16 + l16;
  float bias = bp[co];
#pragma unroll
  for (int r = 0; r < 8; ++r) {
    int p = mt * 16 + r + 8 * half;
    size_t idx = (size_t)(b * CC + co) * NPIX + p;
    out[idx] = x[idx] + acc[r] + bias;
  }
}

extern "C" void kernel_launch(void* const* d_in, const int* in_sizes, int n_in,
                              void* d_out, int out_size, void* d_ws, size_t ws_size,
                              hipStream_t stream) {
  const float* x     = (const float*)d_in[0];
  const float* gamma = (const float*)d_in[1];
  const float* beta  = (const float*)d_in[2];
  const float* wq    = (const float*)d_in[3];
  const float* bq    = (const float*)d_in[4];
  const float* wk    = (const float*)d_in[5];
  const float* bk    = (const float*)d_in[6];
  const float* wv    = (const float*)d_in[7];
  const float* bv    = (const float*)d_in[8];
  const float* wp    = (const float*)d_in[9];
  const float* bp    = (const float*)d_in[10];
  float* out = (float*)d_out;

  char* ws = (char*)d_ws;
  _Float16* w16 = (_Float16*)ws;                                   // 2 MB
  _Float16* hn  = (_Float16*)(ws + (size_t)(2u << 20));            // 32 MB (reused as attn out)
  _Float16* qb  = (_Float16*)(ws + (size_t)(34u << 20));           // 32 MB
  _Float16* kb  = (_Float16*)(ws + (size_t)(66u << 20));           // 32 MB
  _Float16* vtb = (_Float16*)(ws + (size_t)(98u << 20));           // 32 MB
  _Float16* attn = hn;  // hn's last read is qkv_kernel; safe to reuse

  wconv_kernel<<<4096, 256, 0, stream>>>(wq, wk, wv, wp, w16);
  gn_kernel<<<1024, 256, 0, stream>>>(x, gamma, beta, hn);
  qkv_kernel<<<8192, 256, 0, stream>>>(hn, w16, bq, bk, bv, qb, kb, vtb);
  attn_kernel<<<2048, 256, 99328, stream>>>(qb, kb, vtb, attn);
  proj_kernel<<<8192, 256, 0, stream>>>(attn, w16 + 3 * 262144, bp, x, out);
}
